// AttentionBlock_2662879724160
// MI455X (gfx1250) — matrix operand order repaired and checked by reading, not compiled, not run on hardware
//
#include <hip/hip_runtime.h>

// Problem constants (match reference)
#define Bsz 4
#define Tq  2048
#define Cc  1024
#define Kd  1024
#define Vd  1024

typedef __bf16 bf16;
typedef __attribute__((ext_vector_type(16))) __bf16 v16bf;
typedef __attribute__((ext_vector_type(8)))  __bf16 v8bf;
typedef __attribute__((ext_vector_type(8)))  float  v8f;
typedef __attribute__((ext_vector_type(4)))  int    v4i;

// GEMM tiling
#define BM 128
#define BN 128
#define BK 32
#define LDSP 40   // padded LDS row pitch (elements); 40*2=80 bytes, 16B aligned

// ---------------------------------------------------------------------------
// Async global->LDS staging (CDNA5 GLOBAL_LOAD_ASYNC_TO_LDS_B128, ASYNCcnt)
// ---------------------------------------------------------------------------
#if defined(__has_builtin)
#  if __has_builtin(__builtin_amdgcn_global_load_async_to_lds_b128)
#    define HAVE_ASYNC_LDS 1
#  endif
#endif
#ifndef HAVE_ASYNC_LDS
#  define HAVE_ASYNC_LDS 0
#endif

#if HAVE_ASYNC_LDS && defined(__has_builtin)
#  if __has_builtin(__builtin_amdgcn_s_wait_asynccnt)
#    define WAIT_ASYNC() __builtin_amdgcn_s_wait_asynccnt(0)
#  else
#    define WAIT_ASYNC() asm volatile("s_wait_asynccnt 0x0" ::: "memory")
#  endif
#else
#  define WAIT_ASYNC() ((void)0)
#endif

__device__ __forceinline__ void cp16_g2l(bf16* l, const bf16* g)
{
#if HAVE_ASYNC_LDS
    typedef __attribute__((address_space(1))) v4i* gp_t;  // prints as "v4i __device__*"
    typedef __attribute__((address_space(3))) v4i* lp_t;  // 32-bit LDS pointer
    gp_t gp = (gp_t)(unsigned long long)(size_t)g;        // drop const, set AS1
    lp_t lp = (lp_t)(unsigned int)(size_t)l;              // low 32 bits = LDS offset
    __builtin_amdgcn_global_load_async_to_lds_b128(gp, lp, 0, 0);
#else
    *(v8bf*)l = *(const v8bf*)g;
#endif
}

// ---------------------------------------------------------------------------
// WMMA bf16 GEMM:  C[M,N] = A[M,Kin] * B^T-style + (bias)
//  A : [M x Kin] row-major bf16
//  Bm: [N x Kin] row-major bf16 (n-major: weightsT / k-matrix / vT)
//  OUT_MODE 0: bf16, C[row*ldc + col]
//  OUT_MODE 1: fp32, C[row*ldc + c_col_off + col]   (final output)
//  OUT_MODE 2: bf16 transposed, C[col*ldc + row]    (produce vT)
//  CAUSAL_OUT: skip output tiles with n0 > t0 (logits)
//  CAUSAL_K  : inner reduction limited to s < t0+BM (probs @ v)
// ---------------------------------------------------------------------------
template<int OUT_MODE, bool HAS_BIAS, bool CAUSAL_OUT, bool CAUSAL_K>
__global__ __launch_bounds__(256)
void gemm_wmma_bf16(const bf16* __restrict__ A, const bf16* __restrict__ Bm,
                    const float* __restrict__ bias, void* __restrict__ Cout,
                    int lda, int ldb, int ldc, int kIn,
                    long strideA, long strideB, long strideC, int c_col_off)
{
    __shared__ bf16 ldsA[2][BM * LDSP];
    __shared__ bf16 ldsB[2][BN * LDSP];

    const int t0 = blockIdx.y * BM;   // output row tile
    const int n0 = blockIdx.x * BN;   // output col tile
    if (CAUSAL_OUT && n0 > t0) return;

    A  += (long)blockIdx.z * strideA;
    Bm += (long)blockIdx.z * strideB;

    const int tid  = threadIdx.x;
    const int lane = tid & 31;
    const int wv   = tid >> 5;        // wave 0..7
    const int wm   = wv >> 2;         // 0..1  (64-row slice)
    const int wn   = wv & 3;          // 0..3  (32-col slice)
    const int l15  = lane & 15;
    const int h    = (lane >> 4) & 1; // lane half

    // per-thread staging coordinates: 2x16B chunks for A, 2 for B per K-step
    const int srow = tid >> 2;            // 0..63
    const int soff = (tid & 3) * 8;       // 0,8,16,24 (elements)
    const bf16* gA0 = A  + (size_t)(t0 + srow)      * lda + soff;
    const bf16* gA1 = A  + (size_t)(t0 + srow + 64) * lda + soff;
    const bf16* gB0 = Bm + (size_t)(n0 + srow)      * ldb + soff;
    const bf16* gB1 = Bm + (size_t)(n0 + srow + 64) * ldb + soff;
    const int lo0 = srow * LDSP + soff;
    const int lo1 = (srow + 64) * LDSP + soff;

    auto stage = [&](int buf, int kk0) {
        cp16_g2l(&ldsA[buf][lo0], gA0 + kk0);
        cp16_g2l(&ldsA[buf][lo1], gA1 + kk0);
        cp16_g2l(&ldsB[buf][lo0], gB0 + kk0);
        cp16_g2l(&ldsB[buf][lo1], gB1 + kk0);
    };

    v8f acc[4][2];
#pragma unroll
    for (int mt = 0; mt < 4; ++mt)
#pragma unroll
        for (int nt = 0; nt < 2; ++nt)
#pragma unroll
            for (int j = 0; j < 8; ++j) acc[mt][nt][j] = 0.0f;

    const int kTiles = CAUSAL_K ? (t0 + BM) / BK : kIn / BK;

    stage(0, 0);                                  // prologue: fill buffer 0

    for (int kt = 0; kt < kTiles; ++kt) {
        WAIT_ASYNC();                             // this wave's async copies done
        __syncthreads();                          // whole tile visible

        if (kt + 1 < kTiles) stage((kt + 1) & 1, (kt + 1) * BK);

        const bf16* lA = ldsA[kt & 1];
        const bf16* lB = ldsB[kt & 1];

        // ---- gather fragments per CDNA5 16-bit WMMA layouts ----
        v16bf af[4];
#pragma unroll
        for (int mt = 0; mt < 4; ++mt) {
            const bf16* base = lA + (size_t)(wm * 64 + mt * 16 + l15) * LDSP;
            union { v16bf v; v8bf p[2]; } u;
            u.p[0] = *(const v8bf*)(base + h * 8);        // k = h*8 .. h*8+7
            u.p[1] = *(const v8bf*)(base + 16 + h * 8);   // k = 16+h*8 ..
            af[mt] = u.v;
        }
        v16bf bfr[2];
#pragma unroll
        for (int nt = 0; nt < 2; ++nt) {
            const bf16* base = lB + (size_t)(wn * 32 + nt * 16 + l15) * LDSP + h * 8;
            union { v16bf v; v8bf p[2]; } u;
            u.p[0] = *(const v8bf*)(base);                // k = h*8 .. h*8+7
            u.p[1] = *(const v8bf*)(base + 16);           // k = 16+h*8 ..
            bfr[nt] = u.v;
        }

#pragma unroll
        for (int mt = 0; mt < 4; ++mt)
#pragma unroll
            for (int nt = 0; nt < 2; ++nt)
                acc[mt][nt] = __builtin_amdgcn_wmma_f32_16x16x32_bf16(
                    false, af[mt], false, bfr[nt], (short)0, acc[mt][nt], false, false);
    }

    // ---- epilogue: C/D layout VGPR j -> (M = j + h*8, N = lane&15) ----
#pragma unroll
    for (int mt = 0; mt < 4; ++mt) {
#pragma unroll
        for (int nt = 0; nt < 2; ++nt) {
            const int col = n0 + wn * 32 + nt * 16 + l15;
            float bv = 0.0f;
            if constexpr (HAS_BIAS) bv = bias[col];
            const int row0 = t0 + wm * 64 + mt * 16 + h * 8;
            if constexpr (OUT_MODE == 2) {
                // transposed bf16 store: 8 consecutive rows -> contiguous chunk
                bf16* o = (bf16*)Cout;
                v8bf chunk;
#pragma unroll
                for (int j = 0; j < 8; ++j) chunk[j] = (bf16)(acc[mt][nt][j] + bv);
                *(v8bf*)(o + (size_t)col * ldc + row0) = chunk;
            } else {
#pragma unroll
                for (int j = 0; j < 8; ++j) {
                    const int row = row0 + j;
                    const float val = acc[mt][nt][j] + bv;
                    if constexpr (OUT_MODE == 1) {
                        float* o = (float*)Cout + (long)blockIdx.z * strideC;
                        o[(size_t)row * ldc + c_col_off + col] = val;
                    } else {
                        bf16* o = (bf16*)Cout + (long)blockIdx.z * strideC;
                        o[(size_t)row * ldc + col] = (bf16)val;
                    }
                }
            }
        }
    }
}

// ---------------------------------------------------------------------------
// Column softmax over the QUERY axis (reference softmax(axis=1)), in place.
// probs[t,s] = exp((l[t,s]-max_t)/32)/sum_t for t>=s, else 0.
// One block handles 32 key columns of one batch; threads = 32(s) x 8(t).
// ---------------------------------------------------------------------------
__global__ __launch_bounds__(256)
void col_softmax_kernel(bf16* __restrict__ logits)
{
    const int b  = blockIdx.y;
    const int tx = threadIdx.x & 31;
    const int ty = threadIdx.x >> 5;           // 0..7
    const int s  = blockIdx.x * 32 + tx;
    bf16* L = logits + (size_t)b * Tq * Tq;
    __shared__ float red[8][32];

    float m = -3.0e38f;
    for (int t = ty; t < Tq; t += 8)
        if (t >= s) m = fmaxf(m, (float)L[(size_t)t * Tq + s]);
    red[ty][tx] = m;
    __syncthreads();
    if (ty == 0) {
        float mm = red[0][tx];
#pragma unroll
        for (int r = 1; r < 8; ++r) mm = fmaxf(mm, red[r][tx]);
        red[0][tx] = mm;
    }
    __syncthreads();
    const float mcol = red[0][tx];
    __syncthreads();

    const float inv = 0.03125f;               // 1/sqrt(K) = 1/32
    float ssum = 0.0f;
    for (int t = ty; t < Tq; t += 8)
        if (t >= s) ssum += __expf(((float)L[(size_t)t * Tq + s] - mcol) * inv);
    red[ty][tx] = ssum;
    __syncthreads();
    if (ty == 0) {
        float tt = 0.0f;
#pragma unroll
        for (int r = 0; r < 8; ++r) tt += red[r][tx];
        red[0][tx] = tt;
    }
    __syncthreads();
    const float rden = 1.0f / red[0][tx];

    for (int t = ty; t < Tq; t += 8) {
        float p = 0.0f;
        if (t >= s)
            p = __expf(((float)L[(size_t)t * Tq + s] - mcol) * inv) * rden;
        L[(size_t)t * Tq + s] = (bf16)p;      // zero-fill masked region too
    }
}

// ---------------------------------------------------------------------------
__global__ void cvt_f32_bf16(const float* __restrict__ in, bf16* __restrict__ out, int n)
{
    int i = blockIdx.x * blockDim.x + threadIdx.x;
    const int stride = gridDim.x * blockDim.x;
    for (; i < n; i += stride) out[i] = (bf16)in[i];
}

// transpose + convert: in [R][Cn] fp32 -> out [Cn][R] bf16
__global__ __launch_bounds__(256)
void cvt_transpose_bf16(const float* __restrict__ in, bf16* __restrict__ out,
                        int R, int Cn)
{
    __shared__ float tile[32][33];
    const int bx = blockIdx.x * 32;            // col tile (Cn)
    const int by = blockIdx.y * 32;            // row tile (R)
    const int tx = threadIdx.x & 31;
    const int ty = threadIdx.x >> 5;           // 0..7
#pragma unroll
    for (int i = 0; i < 32; i += 8)
        tile[ty + i][tx] = in[(size_t)(by + ty + i) * Cn + bx + tx];
    __syncthreads();
#pragma unroll
    for (int i = 0; i < 32; i += 8)
        out[(size_t)(bx + ty + i) * R + by + tx] = (bf16)tile[tx][ty + i];
}

__global__ void copy_x_kernel(const float* __restrict__ x, float* __restrict__ out)
{
    size_t i = (size_t)blockIdx.x * blockDim.x + threadIdx.x;
    const size_t n = (size_t)Bsz * Tq * Cc;
    const size_t stride = (size_t)gridDim.x * blockDim.x;
    for (; i < n; i += stride) {
        const size_t bt = i >> 10;            // / Cc
        const size_t c  = i & (Cc - 1);
        out[bt * (size_t)(Cc + Vd) + c] = x[i];
    }
}

// ---------------------------------------------------------------------------
extern "C" void kernel_launch(void* const* d_in, const int* in_sizes, int n_in,
                              void* d_out, int out_size, void* d_ws, size_t ws_size,
                              hipStream_t stream)
{
    (void)in_sizes; (void)n_in; (void)out_size; (void)ws_size;

    const float* x  = (const float*)d_in[0];
    const float* Wq = (const float*)d_in[1];
    const float* bq = (const float*)d_in[2];
    const float* Wk = (const float*)d_in[3];
    const float* bk = (const float*)d_in[4];
    const float* Wv = (const float*)d_in[5];
    const float* bv = (const float*)d_in[6];
    float* out = (float*)d_out;

    // workspace layout (bf16), ~102 MB total
    char* p = (char*)d_ws;
    bf16* xb   = (bf16*)p; p += (size_t)Bsz * Tq * Cc * 2;
    bf16* wqT  = (bf16*)p; p += (size_t)Cc * Kd * 2;     // [Kd][Cc]
    bf16* wkT  = (bf16*)p; p += (size_t)Cc * Kd * 2;     // [Kd][Cc]
    bf16* wvT  = (bf16*)p; p += (size_t)Cc * Vd * 2;     // [Vd][Cc]
    bf16* qb   = (bf16*)p; p += (size_t)Bsz * Tq * Kd * 2;
    bf16* kb   = (bf16*)p; p += (size_t)Bsz * Tq * Kd * 2;
    bf16* vT   = (bf16*)p; p += (size_t)Bsz * Tq * Vd * 2;  // [Vd][B*T]
    bf16* lg   = (bf16*)p; p += (size_t)Bsz * Tq * Tq * 2;  // logits -> probs

    // 1) convert x to bf16; transpose-convert weights; copy x into out[:, :C]
    cvt_f32_bf16<<<2048, 256, 0, stream>>>(x, xb, Bsz * Tq * Cc);
    {
        dim3 gT(Kd / 32, Cc / 32, 1);
        cvt_transpose_bf16<<<gT, 256, 0, stream>>>(Wq, wqT, Cc, Kd);
        cvt_transpose_bf16<<<gT, 256, 0, stream>>>(Wk, wkT, Cc, Kd);
        cvt_transpose_bf16<<<gT, 256, 0, stream>>>(Wv, wvT, Cc, Vd);
    }
    copy_x_kernel<<<4096, 256, 0, stream>>>(x, out);

    // 2) projections: q,k normal bf16; v written transposed as vT[Vd][B*T]
    {
        dim3 g(Kd / BN, (Bsz * Tq) / BM, 1);
        gemm_wmma_bf16<0, true, false, false><<<g, 256, 0, stream>>>(
            xb, wqT, bq, qb, Cc, Cc, Kd, Cc, 0, 0, 0, 0);
        gemm_wmma_bf16<0, true, false, false><<<g, 256, 0, stream>>>(
            xb, wkT, bk, kb, Cc, Cc, Kd, Cc, 0, 0, 0, 0);
        gemm_wmma_bf16<2, true, false, false><<<g, 256, 0, stream>>>(
            xb, wvT, bv, vT, Cc, Cc, Bsz * Tq, Cc, 0, 0, 0, 0);
    }

    // 3) logits = q @ k^T per batch (bf16, skip strictly-upper tiles)
    {
        dim3 g(Tq / BN, Tq / BM, Bsz);
        gemm_wmma_bf16<0, false, true, false><<<g, 256, 0, stream>>>(
            qb, kb, nullptr, lg, Kd, Kd, Tq, Kd,
            (long)Tq * Kd, (long)Tq * Kd, (long)Tq * Tq, 0);
    }

    // 4) column softmax (query axis), in place, zeros masked region
    {
        dim3 g(Tq / 32, Bsz, 1);
        col_softmax_kernel<<<g, 256, 0, stream>>>(lg);
    }

    // 5) read = probs @ v (vT is n-major), fp32 into out[:, C:], causal bound
    {
        dim3 g(Vd / BN, Tq / BM, Bsz);
        gemm_wmma_bf16<1, false, false, true><<<g, 256, 0, stream>>>(
            lg, vT, nullptr, out, Tq, Bsz * Tq, Cc + Vd, Tq,
            (long)Tq * Tq, Tq, (long)Tq * (Cc + Vd), Cc);
    }
}
